// MemoryAccessWithUserEmbedding_9002251453170
// MI455X (gfx1250) — matrix operand order repaired and checked by reading, not compiled, not run on hardware
//
#include <hip/hip_runtime.h>
#include <stdint.h>

// ---------- WMMA types ----------
typedef __attribute__((ext_vector_type(16))) __bf16 v16bf;
typedef __attribute__((ext_vector_type(8)))  float  v8f;

union Frag {
  v16bf    v;
  uint16_t u[16];
  uint32_t d[8];
};

__device__ __forceinline__ uint16_t f2bf(float f) {
  uint32_t u = __float_as_uint(f);
  u += 0x7FFFu + ((u >> 16) & 1u);   // round-to-nearest-even
  return (uint16_t)(u >> 16);
}

__device__ __forceinline__ v8f wmma_bf16(const Frag& a, const Frag& b, v8f c) {
  return __builtin_amdgcn_wmma_f32_16x16x32_bf16(
      /*neg_a=*/false, a.v, /*neg_b=*/false, b.v,
      /*c_mod=*/(short)0, c, /*reuse_a=*/false, /*reuse_b=*/false);
}

// A-fragment (16x32 bf16) from a row-major [16 x stride] tile.
// lanes 0-15: row M=lane, K = K0+{0..15}; lanes 16-31: row M=lane-16, K = K0+{16..31}
__device__ __forceinline__ Frag load_afrag(const uint16_t* tile, int stride, int K0, int lane) {
  Frag f;
  int row = lane & 15;
  int c0  = K0 + ((lane >> 4) << 4);
  const uint16_t* p = tile + row * stride + c0;
  *(uint4*)&f.d[0] = *(const uint4*)(p);
  *(uint4*)&f.d[4] = *(const uint4*)(p + 8);
  return f;
}

// B-fragment from pre-packed layout: frag = 32 lanes x 16 bf16, contiguous 32B per lane.
__device__ __forceinline__ Frag load_bfrag(const uint16_t* pack, int fragid, int lane) {
  Frag f;
  const uint16_t* p = pack + (size_t)fragid * 512 + lane * 16;
  *(uint4*)&f.d[0] = *(const uint4*)(p);
  *(uint4*)&f.d[4] = *(const uint4*)(p + 8);
  return f;
}

// ---------- problem constants ----------
#define NB   256
#define NS   128
#define INDIM 384
#define RW   256       // NUM_READS * WORD_SIZE
#define NMEM 512
#define WSZ  64
#define NR   4
#define NSH  16

// ---------- kernel 1: pe = emb[uid] @ W_proc + b_proc ----------
__global__ __launch_bounds__(128) void pe_kernel(const float* __restrict__ emb,
                                                 const float* __restrict__ Wp,
                                                 const float* __restrict__ bp,
                                                 const int*   __restrict__ uid,
                                                 float* __restrict__ pe) {
  int b = blockIdx.x, j = threadIdx.x;
  int u = uid[b];
  const float* e = emb + (size_t)u * 64;
  float acc = bp[j];
#pragma unroll 8
  for (int k = 0; k < 64; ++k) acc += e[k] * Wp[k * 128 + j];
  pe[b * 128 + j] = acc;
}

// ---------- kernel 2: combined_bf16[b,s,384] = concat(x[b,s,:256], pe[b,:128]) ----------
__global__ __launch_bounds__(128) void combined_kernel(const float* __restrict__ x,
                                                       const float* __restrict__ pe,
                                                       uint16_t* __restrict__ comb) {
  int row = blockIdx.x;                 // b*128 + s
  int j   = threadIdx.x;
  int b   = row >> 7;
  const float* xr = x + (size_t)row * 256;
  uint16_t* cr = comb + (size_t)row * INDIM;
  cr[j]       = f2bf(xr[j]);
  cr[j + 128] = f2bf(xr[j + 128]);
  cr[j + 256] = f2bf(pe[b * 128 + j]);
}

// ---------- kernel 3: pack B operands into WMMA B-fragment layout ----------
// Wq frags:   16 shards * 12 ksteps * 16 ntiles  = 3072   (B = Wq[K=384][N=256])
// memT frags: 16 shards *  2 ksteps * 32 ntiles  = 1024   (B[k=w][n=m] = mem[m][w])
// mem  frags: 16 shards * 16 ksteps *  4 ntiles  = 1024   (B[k=m][n=w] = mem[m][w])
__global__ __launch_bounds__(32) void pack_kernel(const float* __restrict__ Wq,
                                                  const float* __restrict__ mem,
                                                  uint16_t* __restrict__ wq_pack,
                                                  uint16_t* __restrict__ memT_pack,
                                                  uint16_t* __restrict__ mem_pack) {
  int f = blockIdx.x;
  int lane = threadIdx.x;
  int nl = lane & 15;
  int kh = (lane >> 4) << 4;            // 0 or 16
  Frag fr;
  uint16_t* dst;
  if (f < 3072) {
    int sid = f / 192, rem = f % 192, ks = rem >> 4, nt = rem & 15;
    int n = nt * 16 + nl;
    const float* W = Wq + (size_t)sid * INDIM * RW;
#pragma unroll
    for (int j = 0; j < 16; ++j) {
      int K = ks * 32 + kh + j;
      fr.u[j] = f2bf(W[(size_t)K * RW + n]);
    }
    dst = wq_pack + (size_t)f * 512 + lane * 16;
  } else if (f < 4096) {
    int f2 = f - 3072;
    int sid = f2 >> 6, rem = f2 & 63, ks = rem >> 5, nt = rem & 31;
    int n = nt * 16 + nl;                              // memory slot m
    const float* Mm = mem + (size_t)sid * NMEM * WSZ;
#pragma unroll
    for (int j = 0; j < 16; ++j) {
      int K = ks * 32 + kh + j;                        // word dim w
      fr.u[j] = f2bf(Mm[(size_t)n * WSZ + K]);         // transpose: B[k=w][n=m]
    }
    dst = memT_pack + (size_t)f2 * 512 + lane * 16;
  } else {
    int f3 = f - 4096;
    int sid = f3 >> 6, rem = f3 & 63, ks = rem >> 2, nt = rem & 3;
    int n = nt * 16 + nl;                              // word dim w
    const float* Mm = mem + (size_t)sid * NMEM * WSZ;
#pragma unroll
    for (int j = 0; j < 16; ++j) {
      int K = ks * 32 + kh + j;                        // memory slot m
      fr.u[j] = f2bf(Mm[(size_t)K * WSZ + n]);         // B[k=m][n=w]
    }
    dst = mem_pack + (size_t)f3 * 512 + lane * 16;
  }
  *(uint4*)(dst)      = *(const uint4*)&fr.d[0];
  *(uint4*)(dst + 8)  = *(const uint4*)&fr.d[4];
}

// ---------- kernel 4: fused q-GEMM -> scores -> softmax -> reads ----------
__global__ __launch_bounds__(128) void main_kernel(const uint16_t* __restrict__ comb,
                                                   const uint16_t* __restrict__ wq_pack,
                                                   const uint16_t* __restrict__ memT_pack,
                                                   const uint16_t* __restrict__ mem_pack,
                                                   const int* __restrict__ uid,
                                                   float* __restrict__ out_reads,
                                                   float* __restrict__ out_state) {
  __shared__ uint16_t a_lds[16 * INDIM]; // 12 KB  combined tile, bf16
  __shared__ uint16_t q_lds[16 * 256];   //  8 KB  q tile, bf16
  __shared__ float    sc_lds[16 * 512];  // 32 KB  scores, fp32
  __shared__ uint16_t w_lds[16 * 512];   // 16 KB  softmax weights, bf16

  int blk  = blockIdx.x;
  int b    = blk >> 3;
  int stile= blk & 7;
  int tid  = threadIdx.x;
  int wave = tid >> 5;
  int lane = tid & 31;
  int sid  = uid[b] & (NSH - 1);

  // ---- Phase 0: stage combined tile [16 x 384] into LDS once (kills 4x redundant global reads)
  {
    const uint4* src = (const uint4*)(comb + ((size_t)b * NS + stile * 16) * INDIM);
    uint4* dst = (uint4*)a_lds;
#pragma unroll
    for (int i = 0; i < (16 * INDIM / 8) / 128; ++i)   // 768 x 16B chunks, 6 per thread
      dst[tid + i * 128] = src[tid + i * 128];
  }
  __syncthreads();

  // ---- Phase 1: q tile [16 x 256] = a_lds[16 x 384] @ Wq[sid]; wave w -> cols [w*64, w*64+64)
  {
    v8f acc0 = {}, acc1 = {}, acc2 = {}, acc3 = {};
    for (int ks = 0; ks < 12; ++ks) {
      Frag a = load_afrag(a_lds, INDIM, ks * 32, lane);
      int fbase = (sid * 12 + ks) * 16 + wave * 4;
      Frag b0 = load_bfrag(wq_pack, fbase + 0, lane);
      Frag b1 = load_bfrag(wq_pack, fbase + 1, lane);
      Frag b2 = load_bfrag(wq_pack, fbase + 2, lane);
      Frag b3 = load_bfrag(wq_pack, fbase + 3, lane);
      acc0 = wmma_bf16(a, b0, acc0);
      acc1 = wmma_bf16(a, b1, acc1);
      acc2 = wmma_bf16(a, b2, acc2);
      acc3 = wmma_bf16(a, b3, acc3);
    }
    int rbase = (lane >> 4) << 3;            // C/D layout: lanes16-31 hold M=8..15
    int col0  = wave * 64 + (lane & 15);
#pragma unroll
    for (int g = 0; g < 8; ++g) {
      q_lds[(g + rbase) * 256 + col0 +  0] = f2bf(acc0[g]);
      q_lds[(g + rbase) * 256 + col0 + 16] = f2bf(acc1[g]);
      q_lds[(g + rbase) * 256 + col0 + 32] = f2bf(acc2[g]);
      q_lds[(g + rbase) * 256 + col0 + 48] = f2bf(acc3[g]);
    }
  }
  __syncthreads();

  // ---- per read-head loop ----
  for (int r = 0; r < NR; ++r) {
    // scores[16 x 512] = q_r[16 x 64] @ memT[sid][64 x 512]; wave w -> ntiles [8w, 8w+8)
    {
      // A-frags invariant across the mt loop: hoist (8x fewer LDS reads)
      Frag a0 = load_afrag(q_lds, 256, r * 64,      lane);
      Frag a1 = load_afrag(q_lds, 256, r * 64 + 32, lane);
      int rbase = (lane >> 4) << 3;
#pragma unroll
      for (int mt = 0; mt < 8; ++mt) {
        int ntile = wave * 8 + mt;
        v8f acc = {};
        Frag bf0 = load_bfrag(memT_pack, (sid * 2 + 0) * 32 + ntile, lane);
        Frag bf1 = load_bfrag(memT_pack, (sid * 2 + 1) * 32 + ntile, lane);
        acc = wmma_bf16(a0, bf0, acc);
        acc = wmma_bf16(a1, bf1, acc);
        int col = ntile * 16 + (lane & 15);
#pragma unroll
        for (int g = 0; g < 8; ++g) sc_lds[(g + rbase) * 512 + col] = acc[g];
      }
    }
    __syncthreads();

    // softmax over 512: 8 lanes per row (aligned 8-lane groups -> shfl_xor 1,2,4), float4 LDS traffic
    {
      int row = tid >> 3;
      int j   = tid & 7;
      float4* srow = (float4*)&sc_lds[row * 512 + j * 64];   // 256B aligned
      float mx = -3.402823466e38f;
#pragma unroll
      for (int c = 0; c < 16; ++c) {
        float4 v = srow[c];
        mx = fmaxf(mx, fmaxf(fmaxf(v.x, v.y), fmaxf(v.z, v.w)));
      }
      mx = fmaxf(mx, __shfl_xor(mx, 1));
      mx = fmaxf(mx, __shfl_xor(mx, 2));
      mx = fmaxf(mx, __shfl_xor(mx, 4));
      float sum = 0.f;
#pragma unroll
      for (int c = 0; c < 16; ++c) {
        float4 v = srow[c];
        v.x = __expf(v.x - mx); v.y = __expf(v.y - mx);
        v.z = __expf(v.z - mx); v.w = __expf(v.w - mx);
        srow[c] = v;
        sum += (v.x + v.y) + (v.z + v.w);
      }
      sum += __shfl_xor(sum, 1);
      sum += __shfl_xor(sum, 2);
      sum += __shfl_xor(sum, 4);
      float inv = __frcp_rn(sum);
      float4* ost = (float4*)(out_state + ((size_t)(stile * 16 + row) * NR + r) * NMEM + j * 64);
#pragma unroll
      for (int c = 0; c < 16; ++c) {
        float4 v = srow[c];
        v.x *= inv; v.y *= inv; v.z *= inv; v.w *= inv;
        ushort4 wv;
        wv.x = f2bf(v.x); wv.y = f2bf(v.y); wv.z = f2bf(v.z); wv.w = f2bf(v.w);
        *(ushort4*)&w_lds[row * 512 + j * 64 + c * 4] = wv;   // 8B aligned
        if (b == NB - 1) ost[c] = v;         // final_state = wts of last sample
      }
    }
    __syncthreads();

    // reads[16 x 64] = wts[16 x 512] @ mem[sid][512 x 64]; wave w -> N cols [w*16, w*16+16)
    // two independent accumulator chains to break the 16-deep WMMA RAW chain
    {
      v8f accA = {}, accB = {};
      for (int ks = 0; ks < 16; ks += 2) {
        Frag aA = load_afrag(w_lds, 512, ks * 32, lane);
        Frag bA = load_bfrag(mem_pack, (sid * 16 + ks) * 4 + wave, lane);
        accA = wmma_bf16(aA, bA, accA);
        Frag aB = load_afrag(w_lds, 512, (ks + 1) * 32, lane);
        Frag bB = load_bfrag(mem_pack, (sid * 16 + ks + 1) * 4 + wave, lane);
        accB = wmma_bf16(aB, bB, accB);
      }
      int rbase = (lane >> 4) << 3;
      int col   = wave * 16 + (lane & 15);
#pragma unroll
      for (int g = 0; g < 8; ++g) {
        size_t s = (size_t)b * NS + stile * 16 + g + rbase;
        out_reads[(s * NR + r) * WSZ + col] = accA[g] + accB[g];
      }
    }
    __syncthreads();   // before next r overwrites sc_lds / w_lds
  }
}

// ---------- launcher ----------
extern "C" void kernel_launch(void* const* d_in, const int* in_sizes, int n_in,
                              void* d_out, int out_size, void* d_ws, size_t ws_size,
                              hipStream_t stream) {
  (void)in_sizes; (void)n_in; (void)out_size; (void)ws_size;
  const float* x    = (const float*)d_in[0];
  const int*   uid  = (const int*)  d_in[1];
  const float* emb  = (const float*)d_in[2];
  const float* Wp   = (const float*)d_in[3];
  const float* bp   = (const float*)d_in[4];
  const float* Wq   = (const float*)d_in[5];
  const float* mem  = (const float*)d_in[6];
  float* out = (float*)d_out;

  char* ws = (char*)d_ws;
  float*    pe        = (float*)   (ws + 0);          //   131,072 B
  uint16_t* comb      = (uint16_t*)(ws + 131072);     // 25,165,824 B
  uint16_t* wq_pack   = (uint16_t*)(ws + 25296896);   //  3,145,728 B
  uint16_t* memT_pack = (uint16_t*)(ws + 28442624);   //  1,048,576 B
  uint16_t* mem_pack  = (uint16_t*)(ws + 29491200);   //  1,048,576 B  (end 30,539,776)

  pe_kernel      <<<NB, 128, 0, stream>>>(emb, Wp, bp, uid, pe);
  combined_kernel<<<NB * NS, 128, 0, stream>>>(x, pe, comb);
  pack_kernel    <<<5120, 32, 0, stream>>>(Wq, mem, wq_pack, memT_pack, mem_pack);
  main_kernel    <<<NB * (NS / 16), 128, 0, stream>>>(comb, wq_pack, memT_pack, mem_pack,
                                                      uid, out, out + (size_t)NB * NS * NR * WSZ);
}
